// AttentionPooling_34815004901954
// MI455X (gfx1250) — matrix operand order repaired and checked by reading, not compiled, not run on hardware
//
#include <hip/hip_runtime.h>
#include <hip/hip_bf16.h>
#include <math.h>

typedef __attribute__((ext_vector_type(16))) __bf16 v16bf;
typedef __attribute__((ext_vector_type(8)))  float  v8f;

#define IN_CH   256
#define HID     128
#define NGRAPH  512
#define SPLIT   4      // sub-ranges per graph in the pooling pass

// ---------------------------------------------------------------------------
// K0: pack W1 (256x128 f32, row-major W1[k][n]) into bf16 WMMA-B fragments.
// Layout: [kt(8)][nt(8)][lane(32)][i(16)] where element i of lane l is
//   W1[kt*32 + (l>>4)*16 + i][nt*16 + (l&15)]
// ---------------------------------------------------------------------------
__global__ void pack_w1_kernel(const float* __restrict__ W1,
                               __bf16* __restrict__ out) {
    int tile = blockIdx.x;                 // 0..63 == kt*8 + nt
    int kt = tile >> 3, nt = tile & 7;
    int l  = threadIdx.x;                  // 0..31
    int col   = nt * 16 + (l & 15);
    int kbase = kt * 32 + (l >> 4) * 16;
    __bf16* dst = out + ((size_t)tile * 32 + l) * 16;
#pragma unroll
    for (int i = 0; i < 16; ++i)
        dst[i] = (__bf16)W1[(size_t)(kbase + i) * HID + col];
}

// ---------------------------------------------------------------------------
// K1: per-node attention score  s = tanh(x@W1 + b1) @ W2 + b2
// One wave32 per 16-node tile; 64 v_wmma_f32_16x16x32_bf16 per wave.
// W1 fragments staged once per block into LDS -> B reads use the ds pipe,
// vmem pipe is left for streaming x at HBM rate.
// ---------------------------------------------------------------------------
__global__ void __launch_bounds__(256)
score_kernel(const float* __restrict__ x,
             const __bf16* __restrict__ w1bf,
             const float* __restrict__ b1,
             const float* __restrict__ W2,
             const float* __restrict__ b2,
             float* __restrict__ s_out,
             int nNodes, int nTiles) {
    __shared__ __bf16 w1s[8 * 8 * 32 * 16];   // 64 KB

    // cooperative stage of packed W1 into LDS (all waves participate)
    {
        const float4* src = (const float4*)w1bf;   // 65536 B = 4096 float4
        float4* dst = (float4*)w1s;
#pragma unroll
        for (int i = threadIdx.x; i < 4096; i += 256)
            dst[i] = src[i];
    }
    __syncthreads();

    int wave = threadIdx.x >> 5;
    int lane = threadIdx.x & 31;
    int tile = blockIdx.x * 8 + wave;
    if (tile >= nTiles) return;            // whole-wave exit after the barrier

    int m0    = tile * 16;
    int r     = lane & 15;
    int khalf = lane >> 4;
    int row   = m0 + r;
    if (row >= nNodes) row = nNodes - 1;   // pad tile by duplicating last row

    const float* xr = x + (size_t)row * IN_CH;

    v8f acc[8];
#pragma unroll
    for (int nt = 0; nt < 8; ++nt)
#pragma unroll
        for (int v = 0; v < 8; ++v) acc[nt][v] = 0.0f;

#pragma unroll
    for (int kt = 0; kt < 8; ++kt) {
        // A k order used here, same as packed W1: elem i (0..15) -> K = khalf*16+i
        const float4* p0 = (const float4*)(xr + kt * 32 + khalf * 16);
        const float4* p1 = (const float4*)(xr + kt * 32 + khalf * 16 + 8);
        float4 u0 = p0[0], u1 = p0[1];
        float4 u2 = p1[0], u3 = p1[1];
        v16bf a;
        a[0]  = (__bf16)u0.x; a[1]  = (__bf16)u0.y; a[2]  = (__bf16)u0.z; a[3]  = (__bf16)u0.w;
        a[4]  = (__bf16)u1.x; a[5]  = (__bf16)u1.y; a[6]  = (__bf16)u1.z; a[7]  = (__bf16)u1.w;
        a[8]  = (__bf16)u2.x; a[9]  = (__bf16)u2.y; a[10] = (__bf16)u2.z; a[11] = (__bf16)u2.w;
        a[12] = (__bf16)u3.x; a[13] = (__bf16)u3.y; a[14] = (__bf16)u3.z; a[15] = (__bf16)u3.w;
#pragma unroll
        for (int nt = 0; nt < 8; ++nt) {
            v16bf bfr = *((const v16bf*)w1s + ((kt * 8 + nt) * 32 + lane));
            acc[nt] = __builtin_amdgcn_wmma_f32_16x16x32_bf16(
                false, a, false, bfr, (short)0, acc[nt], false, false);
        }
    }

    // Epilogue. C/D layout: elem v of lane l = H[M=(l>>4)*8+v][N=nt*16+(l&15)].
    int col = lane & 15;
    float part[8];
#pragma unroll
    for (int v = 0; v < 8; ++v) part[v] = 0.0f;
#pragma unroll
    for (int nt = 0; nt < 8; ++nt) {
        int n = nt * 16 + col;
        float bn = b1[n], wn = W2[n];
#pragma unroll
        for (int v = 0; v < 8; ++v)
            part[v] += tanhf(acc[nt][v] + bn) * wn;
    }
#pragma unroll
    for (int off = 8; off > 0; off >>= 1)
#pragma unroll
        for (int v = 0; v < 8; ++v)
            part[v] += __shfl_xor(part[v], off, 32);

    if (col == 0) {
        float b2v = b2[0];
#pragma unroll
        for (int v = 0; v < 8; ++v) {
            int node = m0 + khalf * 8 + v;
            if (node < nNodes) s_out[node] = part[v] + b2v;
        }
    }
}

// ---------------------------------------------------------------------------
// K2: per-graph max and sum(exp(s - max)); batch sorted -> contiguous ranges.
// ---------------------------------------------------------------------------
__device__ __forceinline__ int lower_bound_i(const int* b, int n, int key) {
    int lo = 0, hi = n;
    while (lo < hi) { int mid = (lo + hi) >> 1; if (b[mid] < key) lo = mid + 1; else hi = mid; }
    return lo;
}

__global__ void segment_stats_kernel(const float* __restrict__ s,
                                     const int* __restrict__ batch,
                                     int nNodes,
                                     float* __restrict__ smax,
                                     float* __restrict__ denom) {
    int g = blockIdx.x, tid = threadIdx.x;
    __shared__ float red[256];
    __shared__ int seb[2];
    if (tid == 0) seb[0] = lower_bound_i(batch, nNodes, g);
    if (tid == 1) seb[1] = lower_bound_i(batch, nNodes, g + 1);
    __syncthreads();
    int start = seb[0], end = seb[1];

    float m = -INFINITY;
    for (int i = start + tid; i < end; i += 256) m = fmaxf(m, s[i]);
    red[tid] = m; __syncthreads();
    for (int off = 128; off > 0; off >>= 1) {
        if (tid < off) red[tid] = fmaxf(red[tid], red[tid + off]);
        __syncthreads();
    }
    float mx = red[0]; __syncthreads();

    float sum = 0.0f;
    for (int i = start + tid; i < end; i += 256) sum += expf(s[i] - mx);
    red[tid] = sum; __syncthreads();
    for (int off = 128; off > 0; off >>= 1) {
        if (tid < off) red[tid] += red[tid + off];
        __syncthreads();
    }
    if (tid == 0) { smax[g] = mx; denom[g] = red[0]; }
}

// ---------------------------------------------------------------------------
// K3: partial[gp][c] = sum over sub-range of x[n][c]*w[n].
// Grid = NGRAPH*SPLIT blocks; thread owns 4 channels (float4/b128 loads),
// 4 node-subsets reduced through LDS -> deterministic.
// ---------------------------------------------------------------------------
__global__ void weighted_pool_kernel(const float* __restrict__ x,
                                     const float* __restrict__ s,
                                     const int* __restrict__ batch,
                                     const float* __restrict__ smax,
                                     const float* __restrict__ denom,
                                     float* __restrict__ partial,
                                     int nNodes) {
    int g = blockIdx.x >> 2;       // graph
    int p = blockIdx.x & 3;        // sub-range
    int tid = threadIdx.x;
    __shared__ float wbuf[256];
    __shared__ int seb[2];
    __shared__ float4 red4[256];
    if (tid == 0) seb[0] = lower_bound_i(batch, nNodes, g);
    if (tid == 1) seb[1] = lower_bound_i(batch, nNodes, g + 1);
    __syncthreads();
    int start = seb[0], end = seb[1];
    int len = end - start;
    int p0 = start + (int)(((long long)len * p)       >> 2);
    int p1 = start + (int)(((long long)len * (p + 1)) >> 2);

    float mx = smax[g];
    float dn = denom[g];
    float inv = dn > 0.0f ? 1.0f / dn : 0.0f;

    int cg  = tid & 63;            // float4 channel group: channels 4cg..4cg+3
    int sub = tid >> 6;            // node subset 0..3
    const float4* x4 = (const float4*)x;
    float4 acc = {0.0f, 0.0f, 0.0f, 0.0f};

    for (int cs = p0; cs < p1; cs += 256) {
        int idx = cs + tid;
        wbuf[tid] = (idx < p1) ? expf(s[idx] - mx) * inv : 0.0f;
        __syncthreads();
        int cnt = min(256, p1 - cs);
#pragma unroll 4
        for (int j = sub; j < cnt; j += 4) {
            float4 xv = x4[(size_t)(cs + j) * 64 + cg];
            float w = wbuf[j];
            acc.x += xv.x * w; acc.y += xv.y * w;
            acc.z += xv.z * w; acc.w += xv.w * w;
        }
        __syncthreads();
    }

    red4[tid] = acc;
    __syncthreads();
    if (tid < 64) {
        float4 a = red4[tid], b = red4[tid + 64], c = red4[tid + 128], d = red4[tid + 192];
        float4 r;
        r.x = (a.x + b.x) + (c.x + d.x);
        r.y = (a.y + b.y) + (c.y + d.y);
        r.z = (a.z + b.z) + (c.z + d.z);
        r.w = (a.w + b.w) + (c.w + d.w);
        ((float4*)partial)[(size_t)blockIdx.x * 64 + tid] = r;
    }
}

// K4: out[g][c] = sum_p partial[g*SPLIT+p][c]  (fixed order -> deterministic)
__global__ void final_reduce_kernel(const float* __restrict__ partial,
                                    float* __restrict__ out) {
    int g = blockIdx.x, c = threadIdx.x;
    const float* pg = partial + (size_t)g * SPLIT * IN_CH + c;
    out[(size_t)g * IN_CH + c] =
        (pg[0] + pg[IN_CH]) + (pg[2 * IN_CH] + pg[3 * IN_CH]);
}

// ---------------------------------------------------------------------------
extern "C" void kernel_launch(void* const* d_in, const int* in_sizes, int n_in,
                              void* d_out, int out_size, void* d_ws, size_t ws_size,
                              hipStream_t stream) {
    const float* x     = (const float*)d_in[0];
    const float* W1    = (const float*)d_in[1];
    const float* b1    = (const float*)d_in[2];
    const float* W2    = (const float*)d_in[3];
    const float* b2    = (const float*)d_in[4];
    const int*   batch = (const int*)d_in[5];
    float* out = (float*)d_out;
    int nNodes = in_sizes[5];

    // ws: [0,64K) W1 bf16 | s[nNodes] | smax[512] | denom[512] | partial[2048*256]
    char* ws = (char*)d_ws;
    __bf16* w1bf = (__bf16*)ws;
    float*  s    = (float*)(ws + 65536);
    size_t soff  = 65536 + (((size_t)nNodes * 4 + 255) / 256) * 256;
    float* smax    = (float*)(ws + soff);
    float* denom   = (float*)(ws + soff + 2048);
    float* partial = (float*)(ws + soff + 4096);   // NGRAPH*SPLIT*IN_CH f32 = 2 MB

    pack_w1_kernel<<<64, 32, 0, stream>>>(W1, w1bf);

    int nTiles  = (nNodes + 15) / 16;
    int nBlocks = (nTiles + 7) / 8;        // 8 waves (tiles) per 256-thread block
    score_kernel<<<nBlocks, 256, 0, stream>>>(x, w1bf, b1, W2, b2, s, nNodes, nTiles);

    segment_stats_kernel<<<NGRAPH, 256, 0, stream>>>(s, batch, nNodes, smax, denom);
    weighted_pool_kernel<<<NGRAPH * SPLIT, 256, 0, stream>>>(x, s, batch, smax, denom,
                                                             partial, nNodes);
    final_reduce_kernel<<<NGRAPH, 256, 0, stream>>>(partial, out);
}
